// ZoneoutLSTMStacked_32186484916733
// MI455X (gfx1250) — matrix operand order repaired and checked by reading, not compiled, not run on hardware
//
#include <hip/hip_runtime.h>
#include <hip/hip_bf16.h>

typedef __attribute__((ext_vector_type(16))) __bf16 v16bf;
typedef __attribute__((ext_vector_type(8)))  __bf16 v8bf;
typedef __attribute__((ext_vector_type(8)))  float  v8f;

#define B_   32
#define T_   16
#define L_   2
#define H_   1024
#define K_   (2 * H_)     // 2048: [x | h] concatenated along K
#define N4H  (4 * H_)     // 4096 gate rows

__device__ __forceinline__ float sigmoidf_(float x) { return 1.0f / (1.0f + __expf(-x)); }

// ---------------------------------------------------------------------------
// Convert W_ih (L,4H,H) and W_hh (L,4H,H) fp32 -> Wc (L,4H,2H) bf16,
// with K = [0..1023] from W_ih and [1024..2047] from W_hh (row-major in K).
// ---------------------------------------------------------------------------
__global__ __launch_bounds__(256)
void k_convert_w(const float* __restrict__ Wih, const float* __restrict__ Whh,
                 __bf16* __restrict__ Wc) {
    int i = blockIdx.x * 256 + threadIdx.x;       // total = L_*N4H*K_ = 16,777,216
    int l = i >> 23;                              // N4H*K_ = 8,388,608 = 1<<23
    int r = i & ((1 << 23) - 1);
    int n = r >> 11;                              // K_ = 2048 = 1<<11
    int k = r & (K_ - 1);
    int src = ((l * N4H + n) << 10);              // *H_
    float v = (k < H_) ? Wih[src + k] : Whh[src + (k - H_)];
    Wc[i] = (__bf16)v;
}

// ---------------------------------------------------------------------------
// Per-launch state init: bsum = b_ih + b_hh, hF/cF = h0/c0,
// act[l][parity0].h = h0[l] (bf16), act[0][parity0].x = x (bf16).
// act buffer layout: 4 buffers (l*2 + parity), each (B_, K_) bf16,
// x-half = cols [0,H), h-half = cols [H,2H).
// ---------------------------------------------------------------------------
__global__ __launch_bounds__(256)
void k_prep(const float* __restrict__ x, const float* __restrict__ h0,
            const float* __restrict__ c0, const float* __restrict__ bih,
            const float* __restrict__ bhh, float* __restrict__ hF,
            float* __restrict__ cF, float* __restrict__ bsum,
            __bf16* __restrict__ act) {
    int idx = blockIdx.x * 256 + threadIdx.x;     // 65536 threads
    if (idx < L_ * B_ * H_) {
        float h = h0[idx];
        hF[idx] = h;
        cF[idx] = c0[idx];
        int l = idx >> 15;                        // B_*H_ = 32768
        int m = (idx >> 10) & (B_ - 1);
        int j = idx & (H_ - 1);
        act[(l * 2 + 0) * (B_ * K_) + m * K_ + H_ + j] = (__bf16)h;
    }
    if (idx < B_ * H_) {
        int m = idx >> 10;
        int j = idx & (H_ - 1);
        act[m * K_ + j] = (__bf16)x[idx];         // buffer (l=0, parity=0) x-half
    }
    if (idx < L_ * N4H) {
        bsum[idx] = bih[idx] + bhh[idx];
    }
}

// ---------------------------------------------------------------------------
// Fused GEMM (bf16 WMMA, f32 acc) + bias + LSTM cell + zoneout for one (t,l).
// grid = 128 blocks: blockIdx = (jtile<<1)|mtile.  block = 128 thr = 4 waves,
// wave w computes gate w's 16x16 tile over K=2048 (64 WMMAs).
// ---------------------------------------------------------------------------
__global__ __launch_bounds__(128)
void k_gemm_cell(const __bf16* __restrict__ Wc,    // (4H, 2H) bf16, this layer
                 const float*  __restrict__ bsum,  // (4H) combined bias
                 const __bf16* __restrict__ actR,  // (B, 2H) read activations
                 float* __restrict__ hF, float* __restrict__ cF,   // (B,H)
                 const float* __restrict__ hm, const float* __restrict__ cm,
                 __bf16* __restrict__ actWh,       // write new h -> h-half here
                 __bf16* __restrict__ actWx,       // write new h -> x-half here
                 float*  __restrict__ outp)        // ret slice (stride T*H) or null
{
    const int tid  = threadIdx.x;
    const int lane = tid & 31;
    const int gate = tid >> 5;            // 0..3 : i, f, g, o
    const int jt   = blockIdx.x >> 1;
    const int mt   = blockIdx.x & 1;
    const int j0   = jt * 16;

    const int nloc = lane & 15;
    const int ncol = gate * H_ + j0 + nloc;          // weight row (gate output)
    const int arow = mt * 16 + nloc;                 // batch row
    // ISA VGPR layouts (wave32):
    //   A 16x32 bf16: lanes 0-15 hold K {0..7, 16..23}; lanes 16-31 {8..15, 24..31}
    //   B 32x16 bf16: lane = N column; lanes 0-15 hold K {0..7, 16..23}; lanes 16-31 {8..15, 24..31}
    const int ka_off = (lane < 16) ? 0 : 8;
    const int kb_off = (lane < 16) ? 0 : 8;

    const __bf16* aptr = actR + arow * K_ + ka_off;
    const __bf16* bptr = Wc + (size_t)ncol * K_ + kb_off;

    v8f acc = {};
    #pragma unroll 4
    for (int k0 = 0; k0 < K_; k0 += 32) {
        v8bf a0 = *(const v8bf*)(aptr + k0);          // K k0..k0+7   (+ka_off)
        v8bf a1 = *(const v8bf*)(aptr + k0 + 16);     // K k0+16..+23 (+ka_off)
        v8bf b0 = *(const v8bf*)(bptr + k0);          // K k0..k0+7   (+kb_off)
        v8bf b1 = *(const v8bf*)(bptr + k0 + 16);     // K k0+16..+23 (+kb_off)
        v16bf a, b;
        #pragma unroll
        for (int i = 0; i < 8; ++i) {
            a[i] = a0[i]; a[8 + i] = a1[i];
            b[i] = b0[i]; b[8 + i] = b1[i];
        }
        acc = __builtin_amdgcn_wmma_f32_16x16x32_bf16(
                  false, a, false, b, (short)0, acc, false, false);
    }

    // Stage gate tiles so each thread sees all four gates of one element.
    __shared__ float gbuf[4][16][16];
    {
        const int rbase = (lane < 16) ? 0 : 8;        // C layout: VGPR r -> M=rbase+r
        #pragma unroll
        for (int r = 0; r < 8; ++r) gbuf[gate][rbase + r][nloc] = acc[r];
    }
    __syncthreads();

    #pragma unroll
    for (int p = 0; p < 2; ++p) {
        int q  = tid + p * 128;                       // 0..255 -> 16x16 elements
        int mm = q >> 4, jj = q & 15;
        int m  = mt * 16 + mm;
        int j  = j0 + jj;
        int sij = m * H_ + j;

        float gi = gbuf[0][mm][jj] + bsum[j];
        float gf = gbuf[1][mm][jj] + bsum[H_ + j];
        float gg = gbuf[2][mm][jj] + bsum[2 * H_ + j];
        float go = gbuf[3][mm][jj] + bsum[3 * H_ + j];

        float cprev = cF[sij];
        float hprev = hF[sij];
        float cn = sigmoidf_(gf) * cprev + sigmoidf_(gi) * tanhf(gg);
        float hc = sigmoidf_(go) * tanhf(cn);
        float hmv = hm[sij], cmv = cm[sij];
        float hn = hmv * hprev + (1.0f - hmv) * hc;   // zoneout: keep prev if mask==1
        float cw = cmv * cprev + (1.0f - cmv) * cn;

        hF[sij] = hn;
        cF[sij] = cw;
        actWh[m * K_ + H_ + j] = (__bf16)hn;          // h input for (l, t+1)
        actWx[m * K_ + j]      = (__bf16)hn;          // x input for (l+1,t) / (0,t+1)
        if (outp) outp[m * (T_ * H_) + j] = hn;       // ret[b][t][j]
    }
}

// ---------------------------------------------------------------------------
// Deterministic single-block MSE reduction.
// ---------------------------------------------------------------------------
__global__ __launch_bounds__(1024)
void k_loss(const float* __restrict__ ret, const float* __restrict__ labels,
            float* __restrict__ out_loss) {
    __shared__ float red[1024];
    const int N = B_ * T_ * H_;
    float s = 0.0f;
    for (int i = threadIdx.x; i < N; i += 1024) {
        float d = ret[i] - labels[i];
        s += d * d;
    }
    red[threadIdx.x] = s;
    __syncthreads();
    for (int off = 512; off > 0; off >>= 1) {
        if (threadIdx.x < off) red[threadIdx.x] += red[threadIdx.x + off];
        __syncthreads();
    }
    if (threadIdx.x == 0) out_loss[0] = red[0] / (float)N;
}

// ---------------------------------------------------------------------------
extern "C" void kernel_launch(void* const* d_in, const int* in_sizes, int n_in,
                              void* d_out, int out_size, void* d_ws, size_t ws_size,
                              hipStream_t stream) {
    const float* x      = (const float*)d_in[0];
    const float* h0     = (const float*)d_in[1];
    const float* c0     = (const float*)d_in[2];
    const float* hmask  = (const float*)d_in[3];
    const float* cmask  = (const float*)d_in[4];
    const float* labels = (const float*)d_in[5];
    const float* Wih    = (const float*)d_in[6];
    const float* Whh    = (const float*)d_in[7];
    const float* bih    = (const float*)d_in[8];
    const float* bhh    = (const float*)d_in[9];
    float* out = (float*)d_out;

    char* ws = (char*)d_ws;
    __bf16* Wc  = (__bf16*)ws;  ws += (size_t)L_ * N4H * K_ * sizeof(__bf16); // 32 MB
    float* bsum = (float*)ws;   ws += (size_t)L_ * N4H * sizeof(float);
    float* hF   = (float*)ws;   ws += (size_t)L_ * B_ * H_ * sizeof(float);
    float* cF   = (float*)ws;   ws += (size_t)L_ * B_ * H_ * sizeof(float);
    __bf16* act = (__bf16*)ws;  ws += (size_t)4 * B_ * K_ * sizeof(__bf16);   // 4 ping-pong bufs

    k_convert_w<<<dim3((L_ * N4H * K_) / 256), dim3(256), 0, stream>>>(Wih, Whh, Wc);
    k_prep<<<dim3(256), dim3(256), 0, stream>>>(x, h0, c0, bih, bhh, hF, cF, bsum, act);

    for (int t = 0; t < T_; ++t) {
        for (int l = 0; l < L_; ++l) {
            // parity ping-pong on t avoids intra-launch read/write races on act
            __bf16* actR  = act + (size_t)(l * 2 + (t & 1)) * (B_ * K_);
            __bf16* actWh = act + (size_t)(l * 2 + ((t + 1) & 1)) * (B_ * K_);
            __bf16* actWx = (l + 1 < L_)
                ? act + (size_t)((l + 1) * 2 + (t & 1)) * (B_ * K_)
                : act + (size_t)(((t + 1) & 1)) * (B_ * K_);       // layer 0, next t
            float* outp = (l == L_ - 1) ? (out + t * H_) : nullptr;

            k_gemm_cell<<<dim3(128), dim3(128), 0, stream>>>(
                Wc + (size_t)l * N4H * K_, bsum + l * N4H, actR,
                hF + l * B_ * H_, cF + l * B_ * H_,
                hmask + (size_t)(t * L_ + l) * B_ * H_,
                cmask + (size_t)(t * L_ + l) * B_ * H_,
                actWh, actWx, outp);
        }
    }

    k_loss<<<dim3(1), dim3(1024), 0, stream>>>(out, labels, out + (size_t)B_ * T_ * H_);
}
